// ObjectDetectorROI_28449863369276
// MI455X (gfx1250) — matrix operand order repaired and checked by reading, not compiled, not run on hardware
//
#include <hip/hip_runtime.h>
#include <hip/hip_bf16.h>

typedef __attribute__((ext_vector_type(16))) __bf16 v16bf;
typedef __attribute__((ext_vector_type(8)))  float  v8f;

__device__ __forceinline__ v8f wmma_bf16(v16bf a, v16bf b, v8f c) {
  // D = A(16x32 bf16) * B(32x16 bf16) + C(16x16 f32)
  return __builtin_amdgcn_wmma_f32_16x16x32_bf16(false, a, false, b, (short)0, c,
                                                 false, false);
}

// ---------------------------------------------------------------------------
// Fused 5x5 VALID conv + bias + relu + 2x2 maxpool, implicit-GEMM via WMMA,
// with the workgroup's input tile staged in LDS:
//   block = 256 threads (8 waves) = one (image, pooled-row) strip of 8 x-tiles
//   LDS tile = CIN x 6 rows x 132 cols (2 output rows + 4 halo, 128 x + 4 halo)
// Fill is coalesced global loads; per-lane im2col gathers become ds loads.
// ---------------------------------------------------------------------------
template<int CIN, int HW, int COUT, int NCHUNK>
__global__ __launch_bounds__(256) void conv5_pool_kernel(
    const float* __restrict__ in, const float* __restrict__ w,
    const float* __restrict__ bias, float* __restrict__ out, size_t outImgStride)
{
  constexpr int K     = CIN * 25;        // 75 or 150
  constexpr int NFULL = K / 32;          // statically guard-free chunks
  constexpr int OWP   = HW - 4;          // pre-pool width (412 / 202)
  constexpr int PW    = OWP / 2;         // pooled width   (206 / 101)
  constexpr int NT    = (OWP + 15) / 16; // x-tiles        (26  / 13)
  constexpr int WT    = 8 * 16 + 4;      // LDS tile width incl. halo (132)
  constexpr int LDSN  = CIN * 6 * WT;

  __shared__ float lds[LDSN];

  const int tid  = threadIdx.x;
  const int lane = tid & 31;
  const int wv   = tid >> 5;
  const int hf   = lane >> 4;            // lane half (0/1)
  const int nm   = lane & 15;            // A-row m / B-col n / C-col n
  const int py = blockIdx.y;             // pooled output row
  const int b  = blockIdx.z;
  const int iy0 = 2 * py;
  const int x0  = blockIdx.x * 128;      // strip origin in x

  // ---- cooperative LDS fill (coalesced, x clamped at right edge) ----------
  const float* inb = in + (size_t)b * CIN * HW * HW;
  for (int idx = tid; idx < LDSN; idx += 256) {
    int c  = idx / (6 * WT);
    int rr = (idx / WT) % 6;
    int xx = idx % WT;
    int ix = x0 + xx; if (ix > HW - 1) ix = HW - 1;
    lds[idx] = inb[((size_t)c * HW + iy0 + rr) * HW + ix];
  }

  // ---- B fragments (K x 16 weights), branchless ---------------------------
  v16bf Bf[NCHUNK];
#pragma unroll
  for (int ck = 0; ck < NCHUNK; ++ck) {
#pragma unroll
    for (int j = 0; j < 16; ++j) {
      int v = j >> 1, e = j & 1;
      int k0 = ck * 32 + hf * 16 + 2 * v + e;
      bool ok = (nm < COUT) && (k0 < K);
      float bw = w[ok ? nm * K + k0 : 0];
      Bf[ck][j] = (__bf16)(ok ? bw : 0.f);
    }
  }

  const bool active = (blockIdx.x * 8 + wv) < NT;
  const int  xtile  = active ? (blockIdx.x * 8 + wv) : NT - 1;
  const int  oxl    = (xtile - blockIdx.x * 8) * 16 + nm;  // x within LDS tile

  __syncthreads();

  v8f acc0 = {}, acc1 = {};
#pragma unroll
  for (int ck = 0; ck < NCHUNK; ++ck) {
    // per-lane LDS gather offsets for this chunk's 16 k values
    int  offs[16];
    bool vald[16];
#pragma unroll
    for (int j = 0; j < 16; ++j) {
      int v = j >> 1, e = j & 1;
      int ka = ck * 32 + hf * 16 + 2 * v + e;
      bool ok = (ck < NFULL) ? true : (ka < K);   // folds for full chunks
      int kc  = ok ? ka : 0;
      int c = kc / 25, rem = kc % 25;
      offs[j] = (c * 6 + rem / 5) * WT + rem % 5; // + rowsel*WT + oxl at use
      vald[j] = ok;
    }
#pragma unroll
    for (int rowsel = 0; rowsel < 2; ++rowsel) {
      const int base = rowsel * WT + oxl;
      float raw[16];
#pragma unroll
      for (int j = 0; j < 16; ++j) raw[j] = lds[base + offs[j]];  // ds loads
      v16bf A;
#pragma unroll
      for (int j = 0; j < 16; ++j)
        A[j] = (__bf16)(vald[j] ? raw[j] : 0.f);
      if (rowsel == 0) acc0 = wmma_bf16(A, Bf[ck], acc0);
      else             acc1 = wmma_bf16(A, Bf[ck], acc1);
    }
  }

  // ---- fused bias + relu + 2x2 maxpool ------------------------------------
  // C layout: lane holds channel nm, rows M = r + 8*hf in VGPR r.
  const float bch = (nm < COUT) ? bias[nm] : 0.f;
#pragma unroll
  for (int i = 0; i < 4; ++i) {
    float m0 = fmaxf(acc0[2 * i], acc0[2 * i + 1]);   // pool x pair, row 2py
    float m1 = fmaxf(acc1[2 * i], acc1[2 * i + 1]);   // pool x pair, row 2py+1
    float pv = fmaxf(fmaxf(m0, m1) + bch, 0.f);       // pool y, +bias, relu
    int px = xtile * 8 + hf * 4 + i;
    if (active && nm < COUT && px < PW)
      out[(size_t)b * outImgStride + ((size_t)nm * PW + py) * PW + px] = pv;
  }
}

// ---------------------------------------------------------------------------
// fc1: C[64,64] += A[64,KPAD] * W[64,163216]^T  (split-K WMMA + f32 atomics)
// A rows padded to KPAD (pad zeroed) -> unguarded b64 loads everywhere;
// W tail chunk uses clamped indices (A pad zero makes the value irrelevant).
// ---------------------------------------------------------------------------
__global__ __launch_bounds__(256) void fc1_kernel(
    const float* __restrict__ x2, const float* __restrict__ w,
    float* __restrict__ cpart)
{
  constexpr int KTOT = 163216, KPAD = 163232, KS = 2048, NSPLIT = 80;
  const int wid = blockIdx.x * 8 + (threadIdx.x >> 5);
  if (wid >= 16 * NSPLIT) return;
  const int lane = threadIdx.x & 31, hf = lane >> 4, nm = lane & 15;
  const int mtile = wid & 3, ntile = (wid >> 2) & 3, ksplit = wid >> 4;

  const float* arow = x2 + (size_t)(mtile * 16 + nm) * KPAD;
  const float* brow = w  + (size_t)(ntile * 16 + nm) * KTOT;

  v8f acc = {};
  const int kbase = ksplit * KS;
  int nfull = (KTOT - kbase) >> 5;
  if (nfull > KS / 32) nfull = KS / 32;

  for (int c = 0; c < nfull; ++c) {
    const int k0 = kbase + c * 32;
    __builtin_prefetch(brow + k0 + 1024, 0, 1);     // stream weight lines
    float2 av[8], bv[8];
#pragma unroll
    for (int v = 0; v < 8; ++v) {
      int ka = k0 + ((v & 4) ? 16 : 0) + hf * 8 + 2 * (v & 3);
      av[v] = *(const float2*)(arow + ka);          // global_load_b64
      int kb = k0 + ((v & 4) ? 16 : 0) + hf * 8 + 2 * (v & 3);
      bv[v] = *(const float2*)(brow + kb);          // global_load_b64
    }
    v16bf A, B;
#pragma unroll
    for (int v = 0; v < 8; ++v) {
      A[2 * v] = (__bf16)av[v].x;  A[2 * v + 1] = (__bf16)av[v].y;
      B[2 * v] = (__bf16)bv[v].x;  B[2 * v + 1] = (__bf16)bv[v].y;
    }
    acc = wmma_bf16(A, B, acc);
  }

  // tail chunk (only the last split reaches it: 163216 % 32 == 16)
  const int k0t = kbase + nfull * 32;
  if (nfull < KS / 32 && k0t < KTOT) {
    v16bf A, B;
#pragma unroll
    for (int v = 0; v < 8; ++v) {
      int ka = k0t + ((v & 4) ? 16 : 0) + hf * 8 + 2 * (v & 3);
      float2 a2 = *(const float2*)(arow + ka);      // pad region holds zeros
      A[2 * v] = (__bf16)a2.x;  A[2 * v + 1] = (__bf16)a2.y;
      int kb = k0t + ((v & 4) ? 16 : 0) + hf * 8 + 2 * (v & 3);
      float b0 = brow[(kb     < KTOT) ? kb     : 0]; // clamp: A==0 beyond KTOT
      float b1 = brow[(kb + 1 < KTOT) ? kb + 1 : 0];
      B[2 * v] = (__bf16)b0;  B[2 * v + 1] = (__bf16)b1;
    }
    acc = wmma_bf16(A, B, acc);
  }

#pragma unroll
  for (int r = 0; r < 8; ++r) {
    int m = mtile * 16 + r + 8 * hf;
    __hip_atomic_fetch_add(&cpart[m * 64 + ntile * 16 + nm], acc[r],
                           __ATOMIC_RELAXED, __HIP_MEMORY_SCOPE_AGENT);
  }
}

__global__ void zero_kernel(float* __restrict__ p, int n) {
  int i = blockIdx.x * blockDim.x + threadIdx.x;
  if (i < n) p[i] = 0.f;
}

// zero the 16-float K-pad of each x2 row (64 rows)
__global__ void padzero_kernel(float* __restrict__ x2) {
  int i = blockIdx.x * blockDim.x + threadIdx.x;
  if (i < 64 * 16)
    x2[(size_t)(i >> 4) * 163232 + 163216 + (i & 15)] = 0.f;
}

// ---------------------------------------------------------------------------
// Decode boxes -> per-roi params {batch, x1, y1, bw, bh} (reference semantics)
// ---------------------------------------------------------------------------
__global__ void box_kernel(const float* __restrict__ cpart,
                           const float* __restrict__ fc1_b,
                           float* __restrict__ rp)
{
  int n = blockIdx.x * blockDim.x + threadIdx.x;
  if (n >= 1024) return;
  const int i = n >> 4, jj = n & 15;
  float v[4];
#pragma unroll
  for (int q = 0; q < 4; ++q) {
    int col = q * 16 + jj;                       // reshape(B,4,16).transpose
    v[q] = fmaxf(cpart[i * 64 + col] + fc1_b[col], 0.f);
  }
  float x1 = fminf(fmaxf(rintf(v[0] * 0.5f), 0.f), 415.f);
  float y1 = fminf(fmaxf(rintf(v[1] * 0.5f), 0.f), 415.f);
  float x2 = fminf(fmaxf(rintf(v[2] * 0.5f), 0.f), 415.f);
  float y2 = fminf(fmaxf(rintf(v[3] * 0.5f), 0.f), 415.f);
  rp[n * 8 + 0] = (float)i;
  rp[n * 8 + 1] = x1;
  rp[n * 8 + 2] = y1;
  rp[n * 8 + 3] = fmaxf(x2 - x1 + 1.f, 1.f) * (1.f / 16.f);
  rp[n * 8 + 4] = fmaxf(y2 - y1 + 1.f, 1.f) * (1.f / 16.f);
}

// ---------------------------------------------------------------------------
// RoI max-pool: one block per roi (gather + max, data-dependent windows)
// ---------------------------------------------------------------------------
__global__ __launch_bounds__(256) void roipool_kernel(
    const float* __restrict__ img, const float* __restrict__ rp,
    float* __restrict__ rois)
{
  const int n = blockIdx.x;
  const int b = (int)rp[n * 8 + 0];
  const float x1 = rp[n * 8 + 1], y1 = rp[n * 8 + 2];
  const float bw = rp[n * 8 + 3], bh = rp[n * 8 + 4];
  for (int idx = threadIdx.x; idx < 3 * 256; idx += 256) {
    const int c = idx >> 8, ph = (idx >> 4) & 15, pw = idx & 15;
    const int hs = (int)fminf(fmaxf(floorf((float)ph * bh) + y1, 0.f), 416.f);
    const int he = (int)fminf(fmaxf(ceilf((float)(ph + 1) * bh) + y1, 0.f), 416.f);
    const int ws = (int)fminf(fmaxf(floorf((float)pw * bw) + x1, 0.f), 416.f);
    const int we = (int)fminf(fmaxf(ceilf((float)(pw + 1) * bw) + x1, 0.f), 416.f);
    const float* p = img + ((size_t)b * 3 + c) * 416 * 416;
    float val = -1e30f;
    for (int h = hs; h < he; ++h)
      for (int q = ws; q < we; ++q)
        val = fmaxf(val, p[h * 416 + q]);
    rois[((size_t)n * 3 + c) * 256 + ph * 16 + pw] =
        (he > hs && we > ws) ? val : 0.f;
  }
}

// ---------------------------------------------------------------------------
// roi_conv (4x4 VALID) + relu -> bf16 [1024, 192] (K padded with zeros)
// ---------------------------------------------------------------------------
__global__ __launch_bounds__(192) void roiconv_kernel(
    const float* __restrict__ rois, const float* __restrict__ w,
    const float* __restrict__ bias, __bf16* __restrict__ y)
{
  const int n = blockIdx.x, t = threadIdx.x;
  if (t >= 169) { y[(size_t)n * 192 + t] = (__bf16)0.f; return; }
  const int oy = t / 13, ox = t % 13;
  float acc = bias[0];
  const float* r0 = rois + (size_t)n * 3 * 256;
#pragma unroll
  for (int c = 0; c < 3; ++c)
#pragma unroll
    for (int r = 0; r < 4; ++r)
#pragma unroll
      for (int s = 0; s < 4; ++s)
        acc += r0[c * 256 + (oy + r) * 16 + ox + s] * w[(c * 4 + r) * 4 + s];
  y[(size_t)n * 192 + t] = (__bf16)fmaxf(acc, 0.f);
}

// ---------------------------------------------------------------------------
// fc2: out[1024,64] = relu(Y[1024,192] * W[64,169]^T + b)   (WMMA)
// ---------------------------------------------------------------------------
__global__ __launch_bounds__(256) void fc2_kernel(
    const __bf16* __restrict__ y, const float* __restrict__ w,
    const float* __restrict__ bias, float* __restrict__ out)
{
  const int wid = blockIdx.x * 8 + (threadIdx.x >> 5);  // 0..255
  const int lane = threadIdx.x & 31, hf = lane >> 4, nm = lane & 15;
  const int ntile = wid & 3, mtile = wid >> 2;          // 4 x 64 tiles
  const __bf16* arow = y + (size_t)(mtile * 16 + nm) * 192;
  const int col = ntile * 16 + nm;
  const float* brow = w + (size_t)col * 169;

  v8f acc = {};
#pragma unroll
  for (int ck = 0; ck < 6; ++ck) {
    const int k0 = ck * 32;
    v16bf A, B;
#pragma unroll
    for (int v = 0; v < 8; ++v) {
      int ka = k0 + ((v & 4) ? 16 : 0) + hf * 8 + 2 * (v & 3);
      union { unsigned u; __bf16 h[2]; } t;
      t.u = *(const unsigned*)(arow + ka);     // paired bf16 load (pad zeroed)
      A[2 * v] = t.h[0];  A[2 * v + 1] = t.h[1];
      int kb = k0 + ((v & 4) ? 16 : 0) + hf * 8 + 2 * (v & 3);
      // chunks 0..4 are statically in-range (kb+1 <= 159 < 169); chunk 5
      // clamps indices (A pad zero makes clamped values irrelevant)
      float b0 = brow[(ck < 5 || kb     < 169) ? kb     : 0];
      float b1 = brow[(ck < 5 || kb + 1 < 169) ? kb + 1 : 0];
      B[2 * v] = (__bf16)b0;  B[2 * v + 1] = (__bf16)b1;
    }
    acc = wmma_bf16(A, B, acc);
  }
  const float bb = bias[col];
#pragma unroll
  for (int r = 0; r < 8; ++r) {
    int m = mtile * 16 + r + 8 * hf;                    // roi index
    out[(size_t)m * 64 + col] = fmaxf(acc[r] + bb, 0.f);
  }
}

// ---------------------------------------------------------------------------
// Workspace layout (float units)
// ---------------------------------------------------------------------------
static constexpr size_t KPAD_X2 = 163232;                   // 163216 + 16 pad
static constexpr size_t N_X1    = 64ull * 6 * 206 * 206;    // 16,295,424
static constexpr size_t N_X2    = 64ull * KPAD_X2;          // 10,446,848
static constexpr size_t OFF_X1  = 0;
static constexpr size_t OFF_X2  = OFF_X1 + N_X1;
static constexpr size_t OFF_CP  = OFF_X2 + N_X2;            // 4096 f
static constexpr size_t OFF_RP  = OFF_CP + 4096;            // 8192 f
static constexpr size_t OFF_ROI = OFF_RP + 8192;            // 786,432 f
static constexpr size_t OFF_Y   = OFF_ROI + 1024ull * 3 * 256; // bf16 [1024,192]

extern "C" void kernel_launch(void* const* d_in, const int* in_sizes, int n_in,
                              void* d_out, int out_size, void* d_ws, size_t ws_size,
                              hipStream_t stream) {
  (void)in_sizes; (void)n_in; (void)out_size; (void)ws_size;
  const float* img        = (const float*)d_in[0];
  const float* conv1_w    = (const float*)d_in[1];
  const float* conv1_b    = (const float*)d_in[2];
  const float* conv2_w    = (const float*)d_in[3];
  const float* conv2_b    = (const float*)d_in[4];
  const float* fc1_w      = (const float*)d_in[5];
  const float* fc1_b      = (const float*)d_in[6];
  const float* roi_conv_w = (const float*)d_in[7];
  const float* roi_conv_b = (const float*)d_in[8];
  const float* fc2_w      = (const float*)d_in[9];
  const float* fc2_b      = (const float*)d_in[10];

  float* wsf  = (float*)d_ws;
  float* x1   = wsf + OFF_X1;
  float* x2   = wsf + OFF_X2;
  float* cp   = wsf + OFF_CP;
  float* rp   = wsf + OFF_RP;
  float* rois = wsf + OFF_ROI;
  __bf16* ybf = (__bf16*)(wsf + OFF_Y);
  float* out  = (float*)d_out;

  // conv1: 26 x-tiles -> 4 blocks of 8 waves; 206 row-pairs; 64 images
  conv5_pool_kernel<3, 416, 6, 3><<<dim3(4, 206, 64), 256, 0, stream>>>(
      img, conv1_w, conv1_b, x1, (size_t)6 * 206 * 206);
  // conv2: 13 x-tiles -> 2 blocks; 101 row-pairs; 64 images; padded row stride
  conv5_pool_kernel<6, 206, 16, 5><<<dim3(2, 101, 64), 256, 0, stream>>>(
      x1, conv2_w, conv2_b, x2, KPAD_X2);
  // fc1: zero partials + K-pad, split-K WMMA with atomics
  zero_kernel<<<16, 256, 0, stream>>>(cp, 4096);
  padzero_kernel<<<4, 256, 0, stream>>>(x2);
  fc1_kernel<<<160, 256, 0, stream>>>(x2, fc1_w, cp);
  // box decode -> roi params
  box_kernel<<<4, 256, 0, stream>>>(cp, fc1_b, rp);
  // roi max-pool
  roipool_kernel<<<1024, 256, 0, stream>>>(img, rp, rois);
  // roi_conv + relu -> bf16 padded activations
  roiconv_kernel<<<1024, 192, 0, stream>>>(rois, roi_conv_w, roi_conv_b, ybf);
  // fc2 WMMA + bias + relu -> d_out [1024,64]
  fc2_kernel<<<32, 256, 0, stream>>>(ybf, fc2_w, fc2_b, out);
}
